// BailingMoeV2SparseMoeBlock_71906342469860
// MI455X (gfx1250) — matrix operand order repaired and checked by reading, not compiled, not run on hardware
//
#include <hip/hip_runtime.h>
#include <hip/hip_bf16.h>
#include <math.h>

#define HDIM 1024      // hidden
#define NEXP 32        // experts
#define NGROUP 4
#define TOPK 4
#define IDIM 512       // routed intermediate
#define I2   (2*IDIM)  // merged gate_up width (1024)
#define INS  1024      // shared intermediate (I*NS)
#define INS2 (2*INS)   // 2048
#define NTOK 2048
#define RSF  2.5f

typedef __attribute__((ext_vector_type(16))) __bf16 v16bf;
typedef __attribute__((ext_vector_type(8)))  float  v8f;
typedef __attribute__((ext_vector_type(4)))  int    v4i;

union FragBF { v16bf v; unsigned int w[8]; unsigned short u[16]; };

// ---- gfx1250 async global->LDS (ASYNCcnt) with synchronous fallback ----
#if defined(__has_builtin)
#  if __has_builtin(__builtin_amdgcn_global_load_async_to_lds_b128) && \
      __has_builtin(__builtin_amdgcn_s_wait_asynccnt)
#    define USE_ASYNC_LDS 1
#  endif
#endif
#ifndef USE_ASYNC_LDS
#  define USE_ASYNC_LDS 0
#endif

__device__ __forceinline__ void copy16_g2l(void* lds_dst, const void* gsrc) {
#if USE_ASYNC_LDS
  __builtin_amdgcn_global_load_async_to_lds_b128(
      (__attribute__((address_space(1))) v4i*)(unsigned long long)gsrc,
      (__attribute__((address_space(3))) v4i*)(unsigned int)(unsigned long long)lds_dst,
      0, 0);
#else
  *(uint4*)lds_dst = *(const uint4*)gsrc;
#endif
}
__device__ __forceinline__ void async_wait() {
#if USE_ASYNC_LDS
  __builtin_amdgcn_s_wait_asynccnt(0);
#endif
}

__device__ __forceinline__ unsigned short f2bf(float f) {
  unsigned int u = __float_as_uint(f);
  u += 0x7FFFu + ((u >> 16) & 1u);          // round-to-nearest-even
  return (unsigned short)(u >> 16);
}
__device__ __forceinline__ float sigmoidf_(float x) { return 1.f / (1.f + __expf(-x)); }

// ---------------------------------------------------------------- converts (vectorized)
__global__ void cvt_bf16(const float4* __restrict__ s, uint2* __restrict__ d, int n4) {
  int i = blockIdx.x * blockDim.x + threadIdx.x;
  if (i < n4) {
    float4 f = s[i];
    unsigned short o[4] = { f2bf(f.x), f2bf(f.y), f2bf(f.z), f2bf(f.w) };
    d[i] = *(uint2*)o;
  }
}

__global__ void zero_counts(int* counts) { if (threadIdx.x < NEXP) counts[threadIdx.x] = 0; }

__global__ void scan_counts(const int* __restrict__ counts, int* __restrict__ offsets) {
  if (threadIdx.x == 0) {
    int a = 0;
    for (int e = 0; e < NEXP; ++e) { offsets[e] = a; a += counts[e]; }
    offsets[NEXP] = a;
  }
}

// ---------------------------------------------------------------- router
// One wave per token; lane == expert (E == wave32 width).
__global__ void __launch_bounds__(128) router_kernel(
    const float* __restrict__ x, const float* __restrict__ Wg,
    const float* __restrict__ bias,
    float* __restrict__ logits_out, int* __restrict__ topk_out,
    int* counts, int* __restrict__ tok_e, float* __restrict__ w_e)
{
  const int t    = (blockIdx.x * blockDim.x + threadIdx.x) >> 5;  // token
  const int lane = threadIdx.x & 31;                              // expert

  const float4* xr = (const float4*)(x  + (size_t)t    * HDIM);
  const float4* wr = (const float4*)(Wg + (size_t)lane * HDIM);
  float acc = 0.f;
  for (int i = 0; i < HDIM / 4; ++i) {
    float4 a = xr[i], b = wr[i];
    acc += a.x * b.x + a.y * b.y + a.z * b.z + a.w * b.w;
  }
  logits_out[(size_t)t * NEXP + lane] = acc;
  float s  = sigmoidf_(acc);
  float sr = s + bias[lane];

  // every lane redundantly gathers all 32 scores (ds_bpermute) and routes
  float sc[NEXP], sb[NEXP];
  for (int e = 0; e < NEXP; ++e) { sc[e] = __shfl(sr, e); sb[e] = __shfl(s, e); }

  float gs[NGROUP];
  for (int g = 0; g < NGROUP; ++g) {
    float m1 = -3.4e38f, m2 = -3.4e38f;
    for (int j = 0; j < 8; ++j) {
      float v = sc[g * 8 + j];
      if (v > m1) { m2 = m1; m1 = v; } else if (v > m2) { m2 = v; }
    }
    gs[g] = m1 + m2;
  }
  int g1 = 0; for (int g = 1; g < NGROUP; ++g) if (gs[g] > gs[g1]) g1 = g;
  gs[g1] = -3.4e38f;
  int g2 = 0; for (int g = 1; g < NGROUP; ++g) if (gs[g] > gs[g2]) g2 = g;
  for (int e = 0; e < NEXP; ++e) { int g = e >> 3; if (g != g1 && g != g2) sc[e] = -3.4e38f; }

  int idx[TOPK]; float wk[TOPK]; float wsum = 0.f;
  for (int k = 0; k < TOPK; ++k) {
    int best = 0;
    for (int e = 1; e < NEXP; ++e) if (sc[e] > sc[best]) best = e;
    idx[k] = best; wk[k] = sb[best]; wsum += sb[best];
    sc[best] = -3.4e38f;
  }
  float inv = RSF / (wsum + 1e-20f);

  if (lane < TOPK) {
    int e = idx[lane];
    topk_out[(size_t)t * TOPK + lane] = e;
    int pos = atomicAdd(&counts[e], 1);
    tok_e[e * NTOK + pos] = t;
    w_e [e * NTOK + pos] = wk[lane] * inv;
  }
}

// ---------------------------------------------------------------- fragment builders
__device__ __forceinline__ void build_fragA(FragBF& A, const unsigned short* lA, int lane) {
  const int M = lane & 15, hi = lane >> 4;
  #pragma unroll
  for (int v = 0; v < 4; ++v) { int k = 2 * v + 8 * hi;            A.w[v] = *(const unsigned int*)&lA[M * 32 + k]; }
  #pragma unroll
  for (int v = 4; v < 8; ++v) { int k = 16 + 2 * (v - 4) + 8 * hi; A.w[v] = *(const unsigned int*)&lA[M * 32 + k]; }
}
__device__ __forceinline__ void build_fragB(FragBF& B, const unsigned short* lBw, int lane) {
  const int N = lane & 15, hi = lane >> 4;
  #pragma unroll
  for (int v = 0; v < 8; ++v) { int k = 2 * (v & 3) + 16 * (v >> 2) + 8 * hi; B.w[v] = *(const unsigned int*)&lBw[N * 32 + k]; }
}

// ---------------------------------------------------------------- routed gate_up + SiLU*mul
// Block: 128 threads = 4 waves. Tile: 16 tokens x 32 i-cols (waves 0/1 gate, 2/3 up).
__global__ void __launch_bounds__(128) gemm_gate_up_routed(
    const unsigned short* __restrict__ xb,
    const unsigned short* __restrict__ Wgu,
    const int* __restrict__ counts, const int* __restrict__ offsets,
    const int* __restrict__ tok_e,
    unsigned short* __restrict__ act)
{
  const int e     = blockIdx.x / (IDIM / 32);
  const int itile = blockIdx.x % (IDIM / 32);
  const int rows  = counts[e];
  const int mt    = blockIdx.y;
  if (mt * 16 >= rows) return;

  const int tid = threadIdx.x, wave = tid >> 5, lane = tid & 31;

  __shared__ unsigned short lA[16 * 32];       // [m][k]
  __shared__ unsigned short lB[4][16 * 32];    // [wave][n*32+k]  (B transposed)
  __shared__ float lC[4][16 * 16];
  __shared__ int toks[16];

  if (tid < 16) {
    int r = mt * 16 + tid;
    toks[tid] = tok_e[e * NTOK + (r < rows ? r : rows - 1)];
  }
  __syncthreads();

  const size_t wbase = (size_t)e * HDIM * I2;
  v8f acc = {0.f, 0.f, 0.f, 0.f, 0.f, 0.f, 0.f, 0.f};

  for (int kt = 0; kt < HDIM / 32; ++kt) {
    if (tid < 64) {                            // async A: 64 x b128 (16B aligned)
      int m = tid >> 2, k0 = (tid & 3) * 8;
      copy16_g2l(&lA[m * 32 + k0], &xb[(size_t)toks[m] * HDIM + kt * 32 + k0]);
    }
    #pragma unroll
    for (int c = 0; c < 2; ++c) {              // B: b128 global loads, transposed ds stores
      int chunk = tid + c * 128;               // 0..255
      int k = chunk >> 3, sub = chunk & 7;
      int col = ((sub >= 4) ? IDIM : 0) + itile * 32 + (sub & 3) * 8;
      uint4 vv = *(const uint4*)&Wgu[wbase + (size_t)(kt * 32 + k) * I2 + col];
      const unsigned short* tt = (const unsigned short*)&vv;
      int w = ((sub >= 4) ? 2 : 0) + ((sub & 3) >> 1);
      int nloc = ((sub & 3) & 1) * 8;
      #pragma unroll
      for (int i = 0; i < 8; ++i) lB[w][(nloc + i) * 32 + k] = tt[i];
    }
    if (kt + 1 < HDIM / 32)
      __builtin_prefetch(&Wgu[wbase + (size_t)((kt + 1) * 32) * I2 + itile * 32], 0, 1);
    async_wait();
    __syncthreads();

    FragBF A, B;
    build_fragA(A, lA, lane);
    build_fragB(B, lB[wave], lane);
    acc = __builtin_amdgcn_wmma_f32_16x16x32_bf16(false, A.v, false, B.v,
                                                  (short)0, acc, false, false);
    __syncthreads();
  }

  {
    const int hi = lane >> 4, N = lane & 15;
    #pragma unroll
    for (int v = 0; v < 8; ++v) lC[wave][(v + 8 * hi) * 16 + N] = acc[v];
  }
  __syncthreads();

  const int obase = offsets[e];
  {                                            // fused SiLU(g)*u -> bf16 act (b64 stores)
    int m = tid >> 3, c0 = (tid & 7) * 4;
    int r = mt * 16 + m;
    if (r < rows) {
      unsigned short o[4];
      #pragma unroll
      for (int i = 0; i < 4; ++i) {
        int c = c0 + i;
        float g = lC[(c >> 4)][m * 16 + (c & 15)];
        float u = lC[2 + (c >> 4)][m * 16 + (c & 15)];
        o[i] = f2bf(g * sigmoidf_(g) * u);
      }
      *(uint2*)&act[(size_t)(obase + r) * IDIM + itile * 32 + c0] = *(uint2*)o;
    }
  }
}

// ---------------------------------------------------------------- routed down + weighted scatter
__global__ void __launch_bounds__(128) gemm_down_routed(
    const unsigned short* __restrict__ act,
    const unsigned short* __restrict__ Wdn,
    const int* __restrict__ counts, const int* __restrict__ offsets,
    const int* __restrict__ tok_e, const float* __restrict__ w_e,
    float* __restrict__ y)
{
  const int e     = blockIdx.x / (HDIM / 64);
  const int htile = blockIdx.x % (HDIM / 64);
  const int rows  = counts[e];
  const int mt    = blockIdx.y;
  if (mt * 16 >= rows) return;

  const int tid = threadIdx.x, wave = tid >> 5, lane = tid & 31;
  const int col = htile * 64 + wave * 16;

  __shared__ unsigned short lA[16 * 32];
  __shared__ unsigned short lB[4][16 * 32];
  __shared__ int   srow[16];
  __shared__ int   stok[16];
  __shared__ float swgt[16];

  if (tid < 16) {
    int r  = mt * 16 + tid;
    int rc = r < rows ? r : rows - 1;
    srow[tid] = offsets[e] + rc;
    stok[tid] = tok_e[e * NTOK + rc];
    swgt[tid] = w_e [e * NTOK + rc];
  }
  __syncthreads();

  const size_t wbase = (size_t)e * IDIM * HDIM;
  v8f acc = {0.f, 0.f, 0.f, 0.f, 0.f, 0.f, 0.f, 0.f};

  for (int kt = 0; kt < IDIM / 32; ++kt) {
    if (tid < 64) {
      int m = tid >> 2, k0 = (tid & 3) * 8;
      copy16_g2l(&lA[m * 32 + k0], &act[(size_t)srow[m] * IDIM + kt * 32 + k0]);
    }
    #pragma unroll
    for (int c = 0; c < 2; ++c) {
      int chunk = tid + c * 128;
      int k = chunk >> 3, sub = chunk & 7;
      uint4 vv = *(const uint4*)&Wdn[wbase + (size_t)(kt * 32 + k) * HDIM + htile * 64 + sub * 8];
      const unsigned short* tt = (const unsigned short*)&vv;
      int w = sub >> 1, nloc = (sub & 1) * 8;
      #pragma unroll
      for (int i = 0; i < 8; ++i) lB[w][(nloc + i) * 32 + k] = tt[i];
    }
    if (kt + 1 < IDIM / 32)
      __builtin_prefetch(&Wdn[wbase + (size_t)((kt + 1) * 32) * HDIM + htile * 64], 0, 1);
    async_wait();
    __syncthreads();

    FragBF A, B;
    build_fragA(A, lA, lane);
    build_fragB(B, lB[wave], lane);
    acc = __builtin_amdgcn_wmma_f32_16x16x32_bf16(false, A.v, false, B.v,
                                                  (short)0, acc, false, false);
    __syncthreads();
  }

  const int hi = lane >> 4, N = lane & 15;
  #pragma unroll
  for (int v = 0; v < 8; ++v) {
    int m = v + 8 * hi;
    int r = mt * 16 + m;
    if (r < rows)
      atomicAdd(&y[(size_t)stok[m] * HDIM + col + N], swgt[m] * acc[v]);
  }
}

// ---------------------------------------------------------------- shared-expert gate_up + SiLU*mul
__global__ void __launch_bounds__(128) gemm_gate_up_shared(
    const unsigned short* __restrict__ xb,
    const unsigned short* __restrict__ Wsgu,
    unsigned short* __restrict__ act_s)
{
  const int itile = blockIdx.x;    // 0..INS/32-1
  const int mt    = blockIdx.y;    // 0..NTOK/16-1
  const int tid = threadIdx.x, wave = tid >> 5, lane = tid & 31;

  __shared__ unsigned short lA[16 * 32];
  __shared__ unsigned short lB[4][16 * 32];
  __shared__ float lC[4][16 * 16];

  v8f acc = {0.f, 0.f, 0.f, 0.f, 0.f, 0.f, 0.f, 0.f};

  for (int kt = 0; kt < HDIM / 32; ++kt) {
    if (tid < 64) {
      int m = tid >> 2, k0 = (tid & 3) * 8;
      copy16_g2l(&lA[m * 32 + k0], &xb[(size_t)(mt * 16 + m) * HDIM + kt * 32 + k0]);
    }
    #pragma unroll
    for (int c = 0; c < 2; ++c) {
      int chunk = tid + c * 128;
      int k = chunk >> 3, sub = chunk & 7;
      int col = ((sub >= 4) ? INS : 0) + itile * 32 + (sub & 3) * 8;
      uint4 vv = *(const uint4*)&Wsgu[(size_t)(kt * 32 + k) * INS2 + col];
      const unsigned short* tt = (const unsigned short*)&vv;
      int w = ((sub >= 4) ? 2 : 0) + ((sub & 3) >> 1);
      int nloc = ((sub & 3) & 1) * 8;
      #pragma unroll
      for (int i = 0; i < 8; ++i) lB[w][(nloc + i) * 32 + k] = tt[i];
    }
    if (kt + 1 < HDIM / 32)
      __builtin_prefetch(&Wsgu[(size_t)((kt + 1) * 32) * INS2 + itile * 32], 0, 1);
    async_wait();
    __syncthreads();

    FragBF A, B;
    build_fragA(A, lA, lane);
    build_fragB(B, lB[wave], lane);
    acc = __builtin_amdgcn_wmma_f32_16x16x32_bf16(false, A.v, false, B.v,
                                                  (short)0, acc, false, false);
    __syncthreads();
  }

  {
    const int hi = lane >> 4, N = lane & 15;
    #pragma unroll
    for (int v = 0; v < 8; ++v) lC[wave][(v + 8 * hi) * 16 + N] = acc[v];
  }
  __syncthreads();

  {
    int m = tid >> 3, c0 = (tid & 7) * 4;
    unsigned short o[4];
    #pragma unroll
    for (int i = 0; i < 4; ++i) {
      int c = c0 + i;
      float g = lC[(c >> 4)][m * 16 + (c & 15)];
      float u = lC[2 + (c >> 4)][m * 16 + (c & 15)];
      o[i] = f2bf(g * sigmoidf_(g) * u);
    }
    *(uint2*)&act_s[(size_t)(mt * 16 + m) * INS + itile * 32 + c0] = *(uint2*)o;
  }
}

// ---------------------------------------------------------------- shared-expert down (initializes y)
__global__ void __launch_bounds__(128) gemm_down_shared(
    const unsigned short* __restrict__ act_s,
    const unsigned short* __restrict__ Wsdn,
    float* __restrict__ y)
{
  const int htile = blockIdx.x;    // 0..HDIM/64-1
  const int mt    = blockIdx.y;
  const int tid = threadIdx.x, wave = tid >> 5, lane = tid & 31;
  const int col = htile * 64 + wave * 16;

  __shared__ unsigned short lA[16 * 32];
  __shared__ unsigned short lB[4][16 * 32];

  v8f acc = {0.f, 0.f, 0.f, 0.f, 0.f, 0.f, 0.f, 0.f};

  for (int kt = 0; kt < INS / 32; ++kt) {
    if (tid < 64) {
      int m = tid >> 2, k0 = (tid & 3) * 8;
      copy16_g2l(&lA[m * 32 + k0], &act_s[(size_t)(mt * 16 + m) * INS + kt * 32 + k0]);
    }
    #pragma unroll
    for (int c = 0; c < 2; ++c) {
      int chunk = tid + c * 128;
      int k = chunk >> 3, sub = chunk & 7;
      uint4 vv = *(const uint4*)&Wsdn[(size_t)(kt * 32 + k) * HDIM + htile * 64 + sub * 8];
      const unsigned short* tt = (const unsigned short*)&vv;
      int w = sub >> 1, nloc = (sub & 1) * 8;
      #pragma unroll
      for (int i = 0; i < 8; ++i) lB[w][(nloc + i) * 32 + k] = tt[i];
    }
    if (kt + 1 < INS / 32)
      __builtin_prefetch(&Wsdn[(size_t)((kt + 1) * 32) * HDIM + htile * 64], 0, 1);
    async_wait();
    __syncthreads();

    FragBF A, B;
    build_fragA(A, lA, lane);
    build_fragB(B, lB[wave], lane);
    acc = __builtin_amdgcn_wmma_f32_16x16x32_bf16(false, A.v, false, B.v,
                                                  (short)0, acc, false, false);
    __syncthreads();
  }

  const int hi = lane >> 4, N = lane & 15;
  #pragma unroll
  for (int v = 0; v < 8; ++v) {
    int m = v + 8 * hi;
    y[(size_t)(mt * 16 + m) * HDIM + col + N] = acc[v];   // plain store: runs before routed scatter
  }
}

// ---------------------------------------------------------------- launch
extern "C" void kernel_launch(void* const* d_in, const int* in_sizes, int n_in,
                              void* d_out, int out_size, void* d_ws, size_t ws_size,
                              hipStream_t stream)
{
  (void)in_sizes; (void)n_in; (void)out_size; (void)ws_size;

  const float* x    = (const float*)d_in[0];
  const float* Wg   = (const float*)d_in[3];
  const float* bias = (const float*)d_in[4];
  const float* Wgu  = (const float*)d_in[5];
  const float* Wdn  = (const float*)d_in[6];
  const float* Wsgu = (const float*)d_in[7];
  const float* Wsdn = (const float*)d_in[8];

  float* y      = (float*)d_out;
  float* logits = y + (size_t)NTOK * HDIM;
  int*   topk   = (int*)(logits + (size_t)NTOK * NEXP);

  char* ws = (char*)d_ws;
  size_t off = 0;
  auto alloc = [&](size_t bytes) -> void* {
    void* p = ws + off;
    off = (off + bytes + 255) & ~(size_t)255;
    return p;
  };
  int*   counts  = (int*)  alloc(NEXP * 4);
  int*   offsets = (int*)  alloc((NEXP + 1) * 4);
  int*   tok_e   = (int*)  alloc((size_t)NEXP * NTOK * 4);
  float* w_e     = (float*)alloc((size_t)NEXP * NTOK * 4);
  unsigned short* xb     = (unsigned short*)alloc((size_t)NTOK * HDIM * 2);
  unsigned short* wgu_b  = (unsigned short*)alloc((size_t)NEXP * HDIM * I2 * 2);
  unsigned short* wdn_b  = (unsigned short*)alloc((size_t)NEXP * IDIM * HDIM * 2);
  unsigned short* wsgu_b = (unsigned short*)alloc((size_t)HDIM * INS2 * 2);
  unsigned short* wsdn_b = (unsigned short*)alloc((size_t)INS * HDIM * 2);
  unsigned short* act    = (unsigned short*)alloc((size_t)NTOK * TOPK * IDIM * 2);
  unsigned short* act_s  = (unsigned short*)alloc((size_t)NTOK * INS * 2);

  auto cvt = [&](const float* s, unsigned short* d, size_t n) {
    int n4 = (int)(n / 4);
    cvt_bf16<<<dim3((unsigned)((n4 + 255) / 256)), dim3(256), 0, stream>>>(
        (const float4*)s, (uint2*)d, n4);
  };
  cvt(x,    xb,     (size_t)NTOK * HDIM);
  cvt(Wgu,  wgu_b,  (size_t)NEXP * HDIM * I2);
  cvt(Wdn,  wdn_b,  (size_t)NEXP * IDIM * HDIM);
  cvt(Wsgu, wsgu_b, (size_t)HDIM * INS2);
  cvt(Wsdn, wsdn_b, (size_t)INS * HDIM);

  zero_counts<<<1, 32, 0, stream>>>(counts);
  router_kernel<<<NTOK / 4, 128, 0, stream>>>(x, Wg, bias, logits, topk, counts, tok_e, w_e);
  scan_counts<<<1, 32, 0, stream>>>(counts, offsets);

  // shared experts first: fully initializes y with plain stores
  gemm_gate_up_shared<<<dim3(INS / 32, NTOK / 16), 128, 0, stream>>>(xb, wsgu_b, act_s);
  gemm_down_shared  <<<dim3(HDIM / 64, NTOK / 16), 128, 0, stream>>>(act_s, wsdn_b, y);

  // routed experts: gather -> GEMM -> weighted atomic scatter into y
  gemm_gate_up_routed<<<dim3(NEXP * (IDIM / 32), NTOK / 16), 128, 0, stream>>>(
      xb, wgu_b, counts, offsets, tok_e, act);
  gemm_down_routed<<<dim3(NEXP * (HDIM / 64), NTOK / 16), 128, 0, stream>>>(
      act, wdn_b, counts, offsets, tok_e, w_e, y);
}
